// MetaModel_71597104824823
// MI455X (gfx1250) — matrix operand order repaired and checked by reading, not compiled, not run on hardware
//
#include <hip/hip_runtime.h>
#include <hip/hip_bf16.h>

// Problem constants (from reference): D=32, H=64, O=1, T=1024, M=8, N=32768, S=2177
#define Dd      32
#define Hh      64
#define Tt      1024
#define Mm      8
#define Nn      32768
#define Rr      576      // 64 base rows + 8*64 rank-update rows
#define RT      36       // Rr / 16 WMMA column tiles
#define SMALL   129      // b1(64) + w2(64) + b2(1) per t
#define SSTRIDE 132
#define YSTRIDE 577      // padded LDS stride for Y[16][576]

typedef __attribute__((ext_vector_type(16))) __bf16 v16bf;
typedef __attribute__((ext_vector_type(8)))  float  v8f;

// ---------------------------------------------------------------------------
// Prep 1: build W_all (576 x 32) = [w1_base+bias ; U_m rows], split into bf16 hi/lo
// ---------------------------------------------------------------------------
__global__ void build_w_kernel(const float* __restrict__ base,
                               const float* __restrict__ bias,
                               const float* __restrict__ meta,
                               __bf16* __restrict__ wh,
                               __bf16* __restrict__ wl) {
    int idx = blockIdx.x * blockDim.x + threadIdx.x;
    if (idx >= Rr * Dd) return;
    int r = idx >> 5;      // row in [0,576)
    int d = idx & 31;      // col in [0,32)
    float v;
    if (r < Hh) {                       // base w1 rows (base + bias)
        int s = r * Dd + d;
        v = base[s] + bias[s];
    } else {                            // U_m rows: meta[(j*32+d)*8 + m]
        int m = (r - Hh) >> 6;
        int j = (r - Hh) & 63;
        v = meta[(j * Dd + d) * Mm + m];
    }
    __bf16 hi = (__bf16)v;
    wh[idx] = hi;
    wl[idx] = (__bf16)(v - (float)hi);
}

// ---------------------------------------------------------------------------
// Prep 2: small per-t state table: b1(64), w2(64), b2(1)  -> 1024 x 132 floats
// ---------------------------------------------------------------------------
__global__ void build_small_kernel(const float* __restrict__ base,
                                   const float* __restrict__ bias,
                                   const float* __restrict__ mesa,
                                   const float* __restrict__ meta,
                                   float* __restrict__ small) {
    int t = blockIdx.x;
    int k = threadIdx.x;
    if (k >= SMALL) return;
    int s = Hh * Dd + k;                // 2048 + k
    float acc = base[s] + bias[s];
#pragma unroll
    for (int m = 0; m < Mm; ++m)
        acc += mesa[m * Tt + t] * meta[s * Mm + m];
    small[t * SSTRIDE + k] = acc;
}

// ---------------------------------------------------------------------------
// Main: per block, 16 samples.  WMMA GEMM  Y(16x576) = X(16x32) * W_all^T,
// bf16 hi/lo split with f32 accumulate, then fused rank-8 mix + ReLU + dot.
// ---------------------------------------------------------------------------
__global__ __launch_bounds__(256) void mesa_main_kernel(
    const float* __restrict__ x,
    const int*   __restrict__ ticker,
    const float* __restrict__ mesa,
    const float* __restrict__ small,
    const __bf16* __restrict__ wh,
    const __bf16* __restrict__ wl,
    float* __restrict__ out) {

    __shared__ __align__(16) float xs[16 * Dd];       // X tile, f32
    __shared__ float ys[16 * YSTRIDE];                // Y tile (16 x 576, padded)
    __shared__ float cb[16 * Mm];                     // per-sample mesa coeffs

    const int tid = threadIdx.x;
    const int n0  = blockIdx.x * 16;

    // ---- stage X tile (16 rows x 32 f32 = 128 float4), coalesced ----
    if (tid < 128) {
        const float4* src = (const float4*)(x + (size_t)n0 * Dd);
        ((float4*)xs)[tid] = src[tid];
    }
    __syncthreads();

    // ---- build A operand per lane (16-bit A 16x32 layout, wave32) ----
    const int lane    = tid & 31;
    const int hihalf  = lane >> 4;        // 0: lanes 0-15, 1: lanes 16-31
    const int arow    = lane & 15;        // M index (sample row)
    v16bf ah, al;
#pragma unroll
    for (int e = 0; e < 16; ++e) {
        // lane<16: K = e (0..15); lane>=16: K = 16 + e
        int k   = e + (hihalf << 4);
        float v = xs[arow * Dd + k];
        __bf16 h = (__bf16)v;
        ah[e] = h;
        al[e] = (__bf16)(v - (float)h);
    }

    // ---- GEMM: 8 waves split the 36 r-tiles; 3 WMMAs per tile (hi/lo) ----
    const int wave = tid >> 5;
    for (int rt = wave; rt < RT; rt += 8) {
        const int r0   = rt * 16;
        // B (32x16) layout: lane holds 16 consecutive d of W_all row (r0 + lane&15)
        const int boff = (r0 + arow) * Dd + (hihalf << 4);
        v16bf bh = *(const v16bf*)(wh + boff);
        v16bf bl = *(const v16bf*)(wl + boff);
        v8f acc = {};
        acc = __builtin_amdgcn_wmma_f32_16x16x32_bf16(false, ah, false, bh, (short)0, acc, false, false);
        acc = __builtin_amdgcn_wmma_f32_16x16x32_bf16(false, ah, false, bl, (short)0, acc, false, false);
        acc = __builtin_amdgcn_wmma_f32_16x16x32_bf16(false, al, false, bh, (short)0, acc, false, false);
        // C/D layout: vgpr v -> row v + 8*hihalf, col = lane&15
#pragma unroll
        for (int vv = 0; vv < 8; ++vv) {
            int row = vv + (hihalf << 3);
            ys[row * YSTRIDE + r0 + arow] = acc[vv];
        }
    }

    // ---- per-sample coefficients c[m] = mesa[m, ticker[n]] ----
    const int nl = tid >> 4;              // local sample 0..15
    const int q  = tid & 15;              // 16 threads per sample
    const int t  = ticker[n0 + nl];
    if (q < Mm) cb[nl * Mm + q] = mesa[q * Tt + t];
    __syncthreads();

    // ---- epilogue: h = relu(Y0 + sum_m c_m*Ym + b1[t]); out = w2[t].h + b2[t]
    const float* sm = small + t * SSTRIDE;
    float po = 0.f;
#pragma unroll
    for (int jj = 0; jj < 4; ++jj) {
        int j = q * 4 + jj;
        float hp = ys[nl * YSTRIDE + j];                       // base part
#pragma unroll
        for (int m = 0; m < Mm; ++m)
            hp += cb[nl * Mm + m] * ys[nl * YSTRIDE + 64 + m * 64 + j];
        hp += sm[j];                                           // + b1[t][j]
        float hr = hp > 0.f ? hp : 0.f;                        // ReLU
        po += sm[64 + j] * hr;                                 // * w2[t][j]
    }
    // reduce 16 partial sums within the wave half (wave32-safe, width 16)
#pragma unroll
    for (int off = 8; off >= 1; off >>= 1)
        po += __shfl_xor(po, off, 16);
    if (q == 0) out[n0 + nl] = po + sm[128];                   // + b2[t]
}

// ---------------------------------------------------------------------------
extern "C" void kernel_launch(void* const* d_in, const int* in_sizes, int n_in,
                              void* d_out, int out_size, void* d_ws, size_t ws_size,
                              hipStream_t stream) {
    const float* x      = (const float*)d_in[0];
    const int*   ticker = (const int*)d_in[1];
    const float* mesa   = (const float*)d_in[2];   // (M, T)
    const float* meta   = (const float*)d_in[3];   // (S, M)
    const float* bias   = (const float*)d_in[4];   // (S,)
    const float* base   = (const float*)d_in[5];   // (S,)
    float* out = (float*)d_out;

    // workspace layout: Wh (36864 B) | Wl (36864 B) | small table (540672 B)
    char*   ws = (char*)d_ws;
    __bf16* wh = (__bf16*)ws;
    __bf16* wl = (__bf16*)(ws + 36864);
    float*  sm = (float*)(ws + 73728);

    build_w_kernel<<<(Rr * Dd + 255) / 256, 256, 0, stream>>>(base, bias, meta, wh, wl);
    build_small_kernel<<<Tt, 160, 0, stream>>>(base, bias, mesa, meta, sm);
    mesa_main_kernel<<<Nn / 16, 256, 0, stream>>>(x, ticker, mesa, sm, wh, wl, out);
}
